// EncoderVis_9096740733666
// MI455X (gfx1250) — matrix operand order repaired and checked by reading, not compiled, not run on hardware
//
#include <hip/hip_runtime.h>
#include <stdint.h>

#define TDIM 500
#define NB 8
#define CHW 6300
#define CHWP 6304       // CHW padded to a multiple of 32 (K-tile)
#define OUTD 128
#define KITERS (CHWP / 32)   // 197 tiles: 0..196 (tile 196 partially past CHW)

typedef __attribute__((ext_vector_type(16))) _Float16 v16h;
typedef __attribute__((ext_vector_type(8)))  float    v8f;

union Frag16 { v16h h; uint32_t u[8]; uint4 q[2]; };
union Acc8   { v8f  v; float    f[8]; };

__device__ __forceinline__ uint32_t pack2h(float a, float b) {
  union { _Float16 h; uint16_t u; } ca, cb;
  ca.h = (_Float16)a; cb.h = (_Float16)b;
  return ((uint32_t)cb.u << 16) | ca.u;
}

// ---- Stage 0: convert W (f32, [OUT][CHW]) -> f16 padded [OUT][CHWP] --------
__global__ void convert_w_kernel(const float* __restrict__ W,
                                 _Float16* __restrict__ W16) {
  int idx = blockIdx.x * blockDim.x + threadIdx.x;
  if (idx >= OUTD * CHWP) return;
  int o = idx / CHWP, k = idx - o * CHWP;
  float v = (k < CHW) ? W[o * CHW + k] : 0.0f;
  W16[idx] = (_Float16)v;
}

// ---- Stage 1: y[n][o][t] = sum_k W[o][k] * x[n][k][t] via f16 WMMA ---------
// Block = 256 thr = 8 waves, handles (n, 16-wide t-tile); wave w -> o-tile.
// B tile (x slab) double-buffered in LDS; one barrier per K-iteration.
// K-guards peeled out of the hot loop (only tile 196 crosses CHW).
__global__ void __launch_bounds__(256)
gemm_kernel(const float* __restrict__ x,
            const _Float16* __restrict__ W16,
            float* __restrict__ y) {
  __shared__ uint32_t lds_b[2][16 * 16];     // [buf][t-col 16][K-pair 16]

  const int n    = blockIdx.y;
  const int t0   = blockIdx.x * 16;
  const int tid  = threadIdx.x;
  const int lane = tid & 31;
  const int o0   = (tid >> 5) * 16;          // wave id * 16

  // staging role: tt fastest within a wave -> 64B-coalesced x loads
  const int tt   = tid & 15;                 // t column 0..15
  const int pair = tid >> 4;                 // K pair 0..15
  const int tcl  = min(t0 + tt, TDIM - 1);   // clamp t (cols >= T never stored)
  const float* xcol = x + (size_t)n * CHW * TDIM + tcl;
  const int ldsw = tt * 16 + pair;           // staging write slot

  // WMMA fragment roles (ISA 7.12.2 layouts)
  const int mrow   = lane & 15;              // A row (o), B col (t), D col (t)
  const int abasek = (lane < 16) ? 0 : 8;    // A: K base per half-wave
  const int bbase  = (lane < 16) ? 0 : 4;    // B: K-pair base per half-wave

  // A-row byte pointer: halves [k0+abasek .. +7] and [k0+16+abasek .. +7]
  // are two 16B-aligned contiguous runs -> explicit b128 loads.
  const char* arow = reinterpret_cast<const char*>(
                         W16 + (size_t)(o0 + mrow) * CHWP) + abasek * 2;

  Acc8 acc;
#pragma unroll
  for (int i = 0; i < 8; i++) acc.f[i] = 0.0f;

  // compute one K-tile: A frags from global W16 (2 x b128), B from LDS (2 x b128)
  auto compute = [&](int k0, int buf) {
    Frag16 a, b;
    a.q[0] = *reinterpret_cast<const uint4*>(arow + k0 * 2);
    a.q[1] = *reinterpret_cast<const uint4*>(arow + k0 * 2 + 32);
    const uint4* bp = reinterpret_cast<const uint4*>(&lds_b[buf][mrow * 16 + bbase]);
    b.q[0] = bp[0];
    b.q[1] = bp[2];
    acc.v = __builtin_amdgcn_wmma_f32_16x16x32_f16(
        false, a.h, false, b.h, (short)0, acc.v, false, false);
  };

  // preload tile 0 (K 0..31 all < CHW: unguarded)
  {
    const float* p = xcol + (size_t)(2 * pair) * TDIM;
    lds_b[0][ldsw] = pack2h(p[0], p[TDIM]);
  }
  __syncthreads();

  // hot loop: tiles 0..194, staging tile it+1 (K <= 6271 < CHW: unguarded)
  const float* xstage = xcol + (size_t)(32 + 2 * pair) * TDIM;
  for (int it = 0; it < KITERS - 2; ++it) {
    const int cur = it & 1;
    compute(it * 32, cur);
    float g0 = xstage[0];
    float g1 = xstage[TDIM];
    xstage += (size_t)32 * TDIM;
    lds_b[cur ^ 1][ldsw] = pack2h(g0, g1);
    __syncthreads();
  }

  // tile 195: compute + guarded staging of tile 196 (K 6272..6303, pad >= CHW)
  {
    int ka = (KITERS - 1) * 32 + 2 * pair;
    int c0 = min(ka, CHW - 1);
    int c1 = min(ka + 1, CHW - 1);
    float l0 = xcol[(size_t)c0 * TDIM];
    float l1 = xcol[(size_t)c1 * TDIM];
    float g0 = (ka     < CHW) ? l0 : 0.0f;
    float g1 = (ka + 1 < CHW) ? l1 : 0.0f;
    compute((KITERS - 2) * 32, (KITERS - 2) & 1);
    lds_b[(KITERS - 1) & 1][ldsw] = pack2h(g0, g1);
    __syncthreads();
  }

  // tile 196: compute only
  compute((KITERS - 1) * 32, (KITERS - 1) & 1);

  // D layout: VGPR v -> M = v (+8 for lanes 16..31), N = lane%16
  const int mhigh = (lane < 16) ? 0 : 8;
  const int t = t0 + mrow;
  if (t < TDIM) {
#pragma unroll
    for (int v = 0; v < 8; v++)
      y[(size_t)(n * OUTD + o0 + mhigh + v) * TDIM + t] = acc.f[v];
  }
}

// ---- Stage 2: fused alpha-PSP recursion + spike/refractory recursion -------
// alpha(x)·W == alpha(x·W): run 2-state alpha scan on y, then spike scan.
__global__ void scan_kernel(const float* __restrict__ y, float* __restrict__ s) {
  int idx = blockIdx.x * blockDim.x + threadIdx.x;   // 0..1023 = n*128+o
  if (idx >= NB * OUTD) return;
  const float* yr = y + (size_t)idx * TDIM;
  float*       sr = s + (size_t)idx * TDIM;

  const float r1 = 0.9048374180359595f;   // exp(-TS/TAU_SR)
  const float c1 = 0.2718281828459045f;   // (TS/TAU_SR)*e
  const float r2 = 0.36787944117144233f;  // exp(-TS/TAU_REF)
  const float c2 = -54.365636569180902f;  // -SCALE_REF*THETA*(TS/TAU_REF)*e
  const float theta = 10.0f;

  float A1 = 0.f, B1 = 0.f, A2 = 0.f, B2 = 0.f;
  for (int t = 0; t < TDIM; t++) {
    float yt = yr[t];
    float u  = A1 + A2;                    // psp output + refractory response
    float B1p = B1 + c1 * yt;              // alpha state update (input y_t)
    A1 = r1 * (A1 + B1p);  B1 = r1 * B1p;
    float sv = (u >= theta) ? 1.0f : 0.0f;
    float B2p = B2 + c2 * sv;              // refractory state update (spike)
    A2 = r2 * (A2 + B2p);  B2 = r2 * B2p;
    sr[t] = sv;
  }
}

extern "C" void kernel_launch(void* const* d_in, const int* in_sizes, int n_in,
                              void* d_out, int out_size, void* d_ws, size_t ws_size,
                              hipStream_t stream) {
  const float* x = (const float*)d_in[0];   // [N,C,H,W,T] f32
  const float* W = (const float*)d_in[1];   // [OUT, CHW]  f32
  float* out = (float*)d_out;               // [N,OUT,1,1,T] f32

  char* ws = (char*)d_ws;
  _Float16* W16 = (_Float16*)ws;                                    // 1.61 MB
  float* y = (float*)(ws + (size_t)OUTD * CHWP * sizeof(_Float16)); // 2.0 MB

  convert_w_kernel<<<(OUTD * CHWP + 255) / 256, 256, 0, stream>>>(W, W16);

  dim3 grid((TDIM + 15) / 16, NB);          // 32 t-tiles x 8 batches
  gemm_kernel<<<grid, 256, 0, stream>>>(x, W16, y);

  scan_kernel<<<8, 128, 0, stream>>>(y, out);
}
